// Gen3Dmol_Classify_52166672777619
// MI455X (gfx1250) — matrix operand order repaired and checked by reading, not compiled, not run on hardware
//
#include <hip/hip_runtime.h>
#include <cstdint>

typedef __bf16 bf16;
typedef bf16  v16bf __attribute__((ext_vector_type(16)));
typedef bf16  v8bf  __attribute__((ext_vector_type(8)));
typedef float v8f   __attribute__((ext_vector_type(8)));

#define BB 8
#define NN 256
#define DD 768
#define HH 8
#define FF 3072
#define KK 128
#define LL 8
#define HDIM 96

__device__ __forceinline__ v8f v8f_zero() {
  v8f z = {0.f,0.f,0.f,0.f,0.f,0.f,0.f,0.f};
  return z;
}

__device__ __forceinline__ float gelu_exact(float x) {
  return 0.5f * x * (1.0f + erff(x * 0.70710678118f));
}

__device__ __forceinline__ v8f wmma_bf16(v16bf a, v16bf b, v8f c) {
  return __builtin_amdgcn_wmma_f32_16x16x32_bf16(false, a, false, b, (short)0, c, false, false);
}

// Fragment-order LDS layout (one v16bf load per lane = whole WMMA operand):
//  A strip (16 rows x 32 k): lane = ((k>>4)<<4)|m,     elem = k&15
//  B tile  (32 k x 16 n):    lane = ((k>>4)<<4)|n,     elem = k&15
__device__ __forceinline__ v8bf cvt8(float4 f0, float4 f1) {
  v8bf r;
  r[0]=(bf16)f0.x; r[1]=(bf16)f0.y; r[2]=(bf16)f0.z; r[3]=(bf16)f0.w;
  r[4]=(bf16)f1.x; r[5]=(bf16)f1.y; r[6]=(bf16)f1.z; r[7]=(bf16)f1.w;
  return r;
}
__device__ __forceinline__ v8bf cvt8s(float4 f0, float4 f1, float sc) {
  v8bf r;
  r[0]=(bf16)(f0.x*sc); r[1]=(bf16)(f0.y*sc); r[2]=(bf16)(f0.z*sc); r[3]=(bf16)(f0.w*sc);
  r[4]=(bf16)(f1.x*sc); r[5]=(bf16)(f1.y*sc); r[6]=(bf16)(f1.z*sc); r[7]=(bf16)(f1.w*sc);
  return r;
}
__device__ __forceinline__ v8bf cvt8a(const float* v) {
  v8bf r;
#pragma unroll
  for (int j = 0; j < 8; ++j) r[j] = (bf16)v[j];
  return r;
}

// ---------------- Fused Gaussian bias kernel -------------------------------
__global__ __launch_bounds__(256)
void gauss_bias_kernel(const float* __restrict__ dist, const int* __restrict__ etype,
                       const float* __restrict__ gmeans, const float* __restrict__ gstds,
                       const float* __restrict__ gmul, const float* __restrict__ gbias,
                       const float* __restrict__ gp1w, const float* __restrict__ gp1b,
                       const float* __restrict__ gp2w, const float* __restrict__ gp2b,
                       float* __restrict__ attn_bias) {
  __shared__ __align__(32) bf16 Gs[4 * 4096];   // 4 k-chunks x 8 strips x 512
  __shared__ __align__(32) bf16 Ws[2][4096];    // double-buffered gp1w chunk
  __shared__ float cmean[KK], cinv[KK], ccoef[KK];
  __shared__ float yrow[128];

  const int b  = blockIdx.y;
  const int p0 = blockIdx.x * 128;
  const int t  = threadIdx.x;
  const int w  = t >> 5;
  const int lane = t & 31;

  for (int k = t; k < KK; k += 256) {
    float sd = fabsf(gstds[k]) + 1e-5f;
    cmean[k] = gmeans[k];
    cinv[k]  = 1.0f / sd;
    ccoef[k] = 1.0f / (2.50662565f * sd);
  }
  for (int r = t; r < 128; r += 256) {
    int p = p0 + r;
    int n = p >> 8, m = p & 255;
    size_t idx = ((size_t)b * NN + n) * NN + m;
    int et = etype[idx];
    yrow[r] = gmul[et] * dist[idx] + gbias[et];
  }
  __syncthreads();

  {  // Gaussian features written directly in fragment order
    int r  = t >> 1;
    int kh = (t & 1) * 64;
    float y = yrow[r];
    int strip = r >> 4, m = r & 15;
#pragma unroll
    for (int o = 0; o < 8; ++o) {
      int kk0 = kh + o * 8;
      v8bf pk8;
#pragma unroll
      for (int j = 0; j < 8; ++j) {
        int kk = kk0 + j;
        float z = (y - cmean[kk]) * cinv[kk];
        pk8[j] = (bf16)(expf(-0.5f * z * z) * ccoef[kk]);
      }
      int chunk = kk0 >> 5, kloc = kk0 & 31;
      int ln = ((kloc >> 4) << 4) | m;
      int pos0 = kloc & 15;
      *(v8bf*)(Gs + chunk * 4096 + strip * 512 + ln * 16 + pos0) = pk8;
    }
  }

  // GEMM1: G[128x128] @ gp1w[128x128], register-pipelined W staging
  float rw[2][8];
  auto gloadW = [&](int k0) {
#pragma unroll
    for (int uu = 0; uu < 2; ++uu) {
      int u = t + uu * 256;
      int oct = u >> 7, c = u & 127;
      int kk0 = k0 + oct * 8;
#pragma unroll
      for (int j = 0; j < 8; ++j) rw[uu][j] = gp1w[(kk0 + j) * KK + c];
    }
  };
  auto lstoreW = [&](int buf) {
#pragma unroll
    for (int uu = 0; uu < 2; ++uu) {
      int u = t + uu * 256;
      int oct = u >> 7, c = u & 127;
      *(v8bf*)(Ws[buf] + (c >> 4) * 512 + (((oct >> 1) << 4) | (c & 15)) * 16 +
               ((oct & 1) << 3)) = cvt8a(rw[uu]);
    }
  };

  v8f acc[8];
#pragma unroll
  for (int i = 0; i < 8; ++i) acc[i] = v8f_zero();

  gloadW(0);
  lstoreW(0);
  __syncthreads();
  int buf = 0;
  for (int chunk = 0; chunk < 4; ++chunk) {
    bool more = chunk < 3;
    if (more) gloadW((chunk + 1) * 32);
    v16bf a = *(const v16bf*)(Gs + chunk * 4096 + w * 512 + lane * 16);
    v16bf bfr[4];
#pragma unroll
    for (int g = 0; g < 2; ++g) {
#pragma unroll
      for (int tc = 0; tc < 4; ++tc)
        bfr[tc] = *(const v16bf*)(Ws[buf] + (g * 4 + tc) * 512 + lane * 16);
#pragma unroll
      for (int tc = 0; tc < 4; ++tc)
        acc[g * 4 + tc] = wmma_bf16(a, bfr[tc], acc[g * 4 + tc]);
    }
    if (more) lstoreW(buf ^ 1);
    __syncthreads();
    buf ^= 1;
  }

  // Epilogue: +gp1b, GELU, rewrite into Gs in fragment order
  const int cn = lane & 15;
  const int rb = (lane >> 4) * 8;
#pragma unroll
  for (int tc = 0; tc < 8; ++tc) {
    int col = tc * 16 + cn;
    float bv = gp1b[col];
    int chunk = col >> 5, kloc = col & 31;
    int lpos = kloc & 15;
    int lhi = (kloc >> 4) << 4;
#pragma unroll
    for (int i = 0; i < 8; ++i) {
      int row = w * 16 + rb + i;
      Gs[chunk * 4096 + w * 512 + (lhi | (row & 15)) * 16 + lpos] =
          (bf16)gelu_exact(acc[tc][i] + bv);
    }
  }
  // stage gp2w (128x8 -> padded 128x16) in fragment order into Ws[0]
  {
    int u = t;
    if (u < 256) {
      int oct = u >> 4, c = u & 15;
      int kk0 = oct * 8;
      v8bf pk8;
#pragma unroll
      for (int j = 0; j < 8; ++j)
        pk8[j] = (c < HH) ? (bf16)gp2w[(kk0 + j) * HH + c] : (bf16)0.0f;
      int chunk = kk0 >> 5, kloc = kk0 & 31;
      int ln = ((kloc >> 4) << 4) | c;
      int pos0 = kloc & 15;
      *(v8bf*)(Ws[0] + chunk * 512 + ln * 16 + pos0) = pk8;
    }
  }
  __syncthreads();

  // GEMM2: h1[128x128] @ w2[128x16]
  v8f acc2 = v8f_zero();
  v16bf a2[4], b2f[4];
#pragma unroll
  for (int chunk = 0; chunk < 4; ++chunk) {
    a2[chunk]  = *(const v16bf*)(Gs + chunk * 4096 + w * 512 + lane * 16);
    b2f[chunk] = *(const v16bf*)(Ws[0] + chunk * 512 + lane * 16);
  }
#pragma unroll
  for (int chunk = 0; chunk < 4; ++chunk)
    acc2 = wmma_bf16(a2[chunk], b2f[chunk], acc2);

  if (cn < HH) {
    float bv = gp2b[cn];
#pragma unroll
    for (int i = 0; i < 8; ++i) {
      int r = w * 16 + rb + i;
      int p = p0 + r;
      int n = p >> 8, m = p & 255;
      attn_bias[(((size_t)b * HH + cn) * NN + n) * NN + m] = acc2[i] + bv;
    }
  }
}

// ---------------- Generic bf16-WMMA GEMM (double-buffered, pipelined) ------
__global__ __launch_bounds__(256)
void gemm_kernel(const float* __restrict__ A, int lda,
                 const float* __restrict__ W, int ldw,
                 const float* __restrict__ bias,
                 const float* __restrict__ resid,
                 float* __restrict__ out, int ldo,
                 int M, int N, int Kd, int act) {
  __shared__ __align__(32) bf16 As[2][4096];
  __shared__ __align__(32) bf16 Bs[2][4096];
  const int t = threadIdx.x;
  const int w = t >> 5;
  const int lane = t & 31;
  const int m0 = blockIdx.y * 128;
  const int n0 = blockIdx.x * 128;

  float4 ra[2][2];
  float  rbv[2][8];
  const float4 f4z = {0.f, 0.f, 0.f, 0.f};

  auto gload = [&](int k0) {
#pragma unroll
    for (int uu = 0; uu < 2; ++uu) {
      int u = t + uu * 256;
      int r = u >> 2, oct = u & 3;
      int gr = m0 + r;
      if (gr < M) {
        const float4* p = (const float4*)(A + (size_t)gr * lda + k0 + oct * 8);
        ra[uu][0] = p[0]; ra[uu][1] = p[1];
      } else { ra[uu][0] = f4z; ra[uu][1] = f4z; }
    }
#pragma unroll
    for (int uu = 0; uu < 2; ++uu) {
      int u = t + uu * 256;
      int oct = u >> 7, c = u & 127;
      int gc = n0 + c;
      int kk0 = k0 + oct * 8;
      if (gc < N) {
#pragma unroll
        for (int j = 0; j < 8; ++j) rbv[uu][j] = W[(size_t)(kk0 + j) * ldw + gc];
      } else {
#pragma unroll
        for (int j = 0; j < 8; ++j) rbv[uu][j] = 0.f;
      }
    }
  };
  auto lstore = [&](int buf) {
#pragma unroll
    for (int uu = 0; uu < 2; ++uu) {
      int u = t + uu * 256;
      int r = u >> 2, oct = u & 3;
      *(v8bf*)(As[buf] + (r >> 4) * 512 + (((oct >> 1) << 4) | (r & 15)) * 16 +
               ((oct & 1) << 3)) = cvt8(ra[uu][0], ra[uu][1]);
    }
#pragma unroll
    for (int uu = 0; uu < 2; ++uu) {
      int u = t + uu * 256;
      int oct = u >> 7, c = u & 127;
      *(v8bf*)(Bs[buf] + (c >> 4) * 512 + (((oct >> 1) << 4) | (c & 15)) * 16 +
               ((oct & 1) << 3)) = cvt8a(rbv[uu]);
    }
  };

  v8f acc[8];
#pragma unroll
  for (int i = 0; i < 8; ++i) acc[i] = v8f_zero();

  gload(0);
  lstore(0);
  __syncthreads();
  int buf = 0;
  for (int k0 = 0; k0 < Kd; k0 += 32) {
    bool more = (k0 + 32 < Kd);
    if (more) gload(k0 + 32);          // HBM/L2 latency hidden behind WMMAs
    v16bf a = *(const v16bf*)(As[buf] + w * 512 + lane * 16);
    v16bf bfr[4];
#pragma unroll
    for (int g = 0; g < 2; ++g) {
#pragma unroll
      for (int tc = 0; tc < 4; ++tc)
        bfr[tc] = *(const v16bf*)(Bs[buf] + (g * 4 + tc) * 512 + lane * 16);
#pragma unroll
      for (int tc = 0; tc < 4; ++tc)
        acc[g * 4 + tc] = wmma_bf16(a, bfr[tc], acc[g * 4 + tc]);
    }
    if (more) lstore(buf ^ 1);
    __syncthreads();
    buf ^= 1;
  }

  const int cn = lane & 15;
  const int rb = (lane >> 4) * 8;
#pragma unroll
  for (int tc = 0; tc < 8; ++tc) {
    int col = n0 + tc * 16 + cn;
    if (col >= N) continue;
    float bv = bias ? bias[col] : 0.0f;
#pragma unroll
    for (int i = 0; i < 8; ++i) {
      int row = m0 + w * 16 + rb + i;
      if (row >= M) continue;
      float v = acc[tc][i] + bv;
      if (act == 1) v = gelu_exact(v);
      if (resid) v += resid[(size_t)row * ldo + col];
      out[(size_t)row * ldo + col] = v;
    }
  }
}

// ---------------- Attention: S = scale*Q @ K^T + bias  (per b,h) -----------
__global__ __launch_bounds__(256)
void attn_score_kernel(const float* __restrict__ pk, const float* __restrict__ attn_bias,
                       float* __restrict__ s, float scale) {
  __shared__ __align__(32) bf16 As[2][4096];
  __shared__ __align__(32) bf16 Bs[2][4096];
  const int bh = blockIdx.z;
  const int b = bh >> 3, h = bh & 7;
  const int i0 = blockIdx.y * 128;
  const int j0 = blockIdx.x * 128;
  const float* base = pk + (size_t)b * NN * DD + h * HDIM;
  const int t = threadIdx.x, w = t >> 5, lane = t & 31;

  float4 ra[2][2], rbq[2][2];
  auto gload = [&](int k0) {
#pragma unroll
    for (int uu = 0; uu < 2; ++uu) {
      int u = t + uu * 256;
      int r = u >> 2, oct = u & 3;
      const float4* p = (const float4*)(base + (size_t)(i0 + r) * DD + k0 + oct * 8);
      ra[uu][0] = p[0]; ra[uu][1] = p[1];
      const float4* q = (const float4*)(base + (size_t)(j0 + r) * DD + k0 + oct * 8);
      rbq[uu][0] = q[0]; rbq[uu][1] = q[1];
    }
  };
  auto lstore = [&](int buf) {
#pragma unroll
    for (int uu = 0; uu < 2; ++uu) {
      int u = t + uu * 256;
      int r = u >> 2, oct = u & 3;
      *(v8bf*)(As[buf] + (r >> 4) * 512 + (((oct >> 1) << 4) | (r & 15)) * 16 +
               ((oct & 1) << 3)) = cvt8s(ra[uu][0], ra[uu][1], scale);
      *(v8bf*)(Bs[buf] + (r >> 4) * 512 + (((oct >> 1) << 4) | (r & 15)) * 16 +
               ((oct & 1) << 3)) = cvt8(rbq[uu][0], rbq[uu][1]);
    }
  };

  v8f acc[8];
#pragma unroll
  for (int i = 0; i < 8; ++i) acc[i] = v8f_zero();

  gload(0);
  lstore(0);
  __syncthreads();
  int buf = 0;
  for (int k0 = 0; k0 < HDIM; k0 += 32) {
    bool more = (k0 + 32 < HDIM);
    if (more) gload(k0 + 32);
    v16bf a = *(const v16bf*)(As[buf] + w * 512 + lane * 16);
    v16bf bfr[4];
#pragma unroll
    for (int g = 0; g < 2; ++g) {
#pragma unroll
      for (int tc = 0; tc < 4; ++tc)
        bfr[tc] = *(const v16bf*)(Bs[buf] + (g * 4 + tc) * 512 + lane * 16);
#pragma unroll
      for (int tc = 0; tc < 4; ++tc)
        acc[g * 4 + tc] = wmma_bf16(a, bfr[tc], acc[g * 4 + tc]);
    }
    if (more) lstore(buf ^ 1);
    __syncthreads();
    buf ^= 1;
  }

  const int cn = lane & 15;
  const int rb = (lane >> 4) * 8;
  const size_t sb = (size_t)bh * NN * NN;
#pragma unroll
  for (int tc = 0; tc < 8; ++tc) {
    int j = j0 + tc * 16 + cn;
#pragma unroll
    for (int i = 0; i < 8; ++i) {
      int ii = i0 + w * 16 + rb + i;
      size_t off = sb + (size_t)ii * NN + j;
      s[off] = acc[tc][i] + attn_bias[off];
    }
  }
}

// ---------------- Softmax over last dim (N=256, one row per block) ---------
__global__ __launch_bounds__(256)
void softmax_kernel(float* __restrict__ s) {
  __shared__ float red[256];
  float* p = s + (size_t)blockIdx.x * NN;
  const int t = threadIdx.x;
  float v = p[t];
  red[t] = v; __syncthreads();
  for (int st = 128; st > 0; st >>= 1) {
    if (t < st) red[t] = fmaxf(red[t], red[t + st]);
    __syncthreads();
  }
  float mx = red[0]; __syncthreads();
  float e = expf(v - mx);
  red[t] = e; __syncthreads();
  for (int st = 128; st > 0; st >>= 1) {
    if (t < st) red[t] += red[t + st];
    __syncthreads();
  }
  p[t] = e * (1.0f / red[0]);
}

// ---------------- O = P @ V  (per b,h; 96 output cols) ---------------------
__global__ __launch_bounds__(256)
void attn_pv_kernel(const float* __restrict__ s, const float* __restrict__ pk,
                    float* __restrict__ o) {
  __shared__ __align__(32) bf16 As[2][4096];
  __shared__ __align__(32) bf16 Bs[2][3072];
  const int bh = blockIdx.y;
  const int b = bh >> 3, h = bh & 7;
  const int i0 = blockIdx.x * 128;
  const float* vbase = pk + (size_t)b * NN * DD + h * HDIM;
  const float* sbase = s + (size_t)bh * NN * NN;
  const int t = threadIdx.x, w = t >> 5, lane = t & 31;

  float4 ra[2][2];
  float  rbv[2][8];
  auto gload = [&](int k0) {
#pragma unroll
    for (int uu = 0; uu < 2; ++uu) {
      int u = t + uu * 256;
      int r = u >> 2, oct = u & 3;
      const float4* p = (const float4*)(sbase + (size_t)(i0 + r) * NN + k0 + oct * 8);
      ra[uu][0] = p[0]; ra[uu][1] = p[1];
    }
#pragma unroll
    for (int uu = 0; uu < 2; ++uu) {
      int u = t + uu * 256;
      if (u < 384) {
        int oct = u / 96;
        int c = u - oct * 96;
        int kk0 = k0 + oct * 8;
#pragma unroll
        for (int j = 0; j < 8; ++j) rbv[uu][j] = vbase[(size_t)(kk0 + j) * DD + c];
      }
    }
  };
  auto lstore = [&](int buf) {
#pragma unroll
    for (int uu = 0; uu < 2; ++uu) {
      int u = t + uu * 256;
      int r = u >> 2, oct = u & 3;
      *(v8bf*)(As[buf] + (r >> 4) * 512 + (((oct >> 1) << 4) | (r & 15)) * 16 +
               ((oct & 1) << 3)) = cvt8(ra[uu][0], ra[uu][1]);
    }
#pragma unroll
    for (int uu = 0; uu < 2; ++uu) {
      int u = t + uu * 256;
      if (u < 384) {
        int oct = u / 96;
        int c = u - oct * 96;
        *(v8bf*)(Bs[buf] + (c >> 4) * 512 + (((oct >> 1) << 4) | (c & 15)) * 16 +
                 ((oct & 1) << 3)) = cvt8a(rbv[uu]);
      }
    }
  };

  v8f acc[6];
#pragma unroll
  for (int i = 0; i < 6; ++i) acc[i] = v8f_zero();

  gload(0);
  lstore(0);
  __syncthreads();
  int buf = 0;
  for (int k0 = 0; k0 < NN; k0 += 32) {
    bool more = (k0 + 32 < NN);
    if (more) gload(k0 + 32);
    v16bf a = *(const v16bf*)(As[buf] + w * 512 + lane * 16);
    v16bf bfr[6];
#pragma unroll
    for (int tc = 0; tc < 6; ++tc)
      bfr[tc] = *(const v16bf*)(Bs[buf] + tc * 512 + lane * 16);
#pragma unroll
    for (int tc = 0; tc < 6; ++tc)
      acc[tc] = wmma_bf16(a, bfr[tc], acc[tc]);
    if (more) lstore(buf ^ 1);
    __syncthreads();
    buf ^= 1;
  }

  const int cn = lane & 15;
  const int rb = (lane >> 4) * 8;
#pragma unroll
  for (int tc = 0; tc < 6; ++tc) {
    int c = tc * 16 + cn;
#pragma unroll
    for (int i = 0; i < 8; ++i) {
      int ii = i0 + w * 16 + rb + i;
      o[((size_t)b * NN + ii) * DD + h * HDIM + c] = acc[tc][i];
    }
  }
}

// ---------------- Embedding gather + LayerNorm -----------------------------
__global__ __launch_bounds__(256)
void gather_embed_kernel(const float* __restrict__ emb, const int* __restrict__ tokens,
                         float* __restrict__ x) {
  size_t row = blockIdx.x;
  int tok = tokens[row];
  const float* src = emb + (size_t)tok * DD;
  float* dst = x + row * DD;
  for (int d = threadIdx.x; d < DD; d += 256) dst[d] = src[d];
}

__global__ __launch_bounds__(256)
void ln_kernel(const float* __restrict__ x, const float* __restrict__ g,
               const float* __restrict__ bta, float* __restrict__ y) {
  __shared__ float red[256];
  size_t row = blockIdx.x;
  const float* xr = x + row * DD;
  const int t = threadIdx.x;
  float s = 0.f;
  for (int d = t; d < DD; d += 256) s += xr[d];
  red[t] = s; __syncthreads();
  for (int st = 128; st > 0; st >>= 1) {
    if (t < st) red[t] += red[t + st];
    __syncthreads();
  }
  float mean = red[0] * (1.0f / DD);
  __syncthreads();
  float v = 0.f;
  for (int d = t; d < DD; d += 256) { float dd = xr[d] - mean; v += dd * dd; }
  red[t] = v; __syncthreads();
  for (int st = 128; st > 0; st >>= 1) {
    if (t < st) red[t] += red[t + st];
    __syncthreads();
  }
  float inv = rsqrtf(red[0] * (1.0f / DD) + 1e-3f);
  for (int d = t; d < DD; d += 256)
    y[row * DD + d] = (xr[d] - mean) * inv * g[d] + bta[d];
}

// ---------------- Host-side orchestration ----------------------------------
extern "C" void kernel_launch(void* const* d_in, const int* in_sizes, int n_in,
                              void* d_out, int out_size, void* d_ws, size_t ws_size,
                              hipStream_t stream) {
  (void)in_sizes; (void)n_in; (void)out_size; (void)ws_size;
  const float* dist    = (const float*)d_in[0];
  const float* tok_emb = (const float*)d_in[1];
  const float* gmeans  = (const float*)d_in[2];
  const float* gstds   = (const float*)d_in[3];
  const float* gmul    = (const float*)d_in[4];
  const float* gbias   = (const float*)d_in[5];
  const float* gp1w    = (const float*)d_in[6];
  const float* gp1b    = (const float*)d_in[7];
  const float* gp2w    = (const float*)d_in[8];
  const float* gp2b    = (const float*)d_in[9];
  const float* emb_g   = (const float*)d_in[10];
  const float* emb_b   = (const float*)d_in[11];
  const float* Wq      = (const float*)d_in[12];
  const float* bq      = (const float*)d_in[13];
  const float* Wo      = (const float*)d_in[14];
  const float* bo      = (const float*)d_in[15];
  const float* ln1g    = (const float*)d_in[16];
  const float* ln1b    = (const float*)d_in[17];
  const float* W1      = (const float*)d_in[18];
  const float* b1      = (const float*)d_in[19];
  const float* W2      = (const float*)d_in[20];
  const float* b2      = (const float*)d_in[21];
  const float* ln2g    = (const float*)d_in[22];
  const float* ln2b    = (const float*)d_in[23];
  const float* fing    = (const float*)d_in[24];
  const float* finb    = (const float*)d_in[25];
  const float* cw1     = (const float*)d_in[26];
  const float* cb1     = (const float*)d_in[27];
  const float* cw2     = (const float*)d_in[28];
  const float* cb2     = (const float*)d_in[29];
  const int*   tokens  = (const int*)d_in[30];
  const int*   etype   = (const int*)d_in[31];
  float* out = (float*)d_out;

  float* ws = (float*)d_ws;
  float* attn_bias = ws; ws += (size_t)BB * HH * NN * NN;
  float* sbuf      = ws; ws += (size_t)BB * HH * NN * NN;
  float* xbuf      = ws; ws += (size_t)BB * NN * DD;
  float* hbuf      = ws; ws += (size_t)BB * NN * DD;
  float* pbuf      = ws; ws += (size_t)BB * NN * DD;
  float* obuf      = ws; ws += (size_t)BB * NN * DD;
  float* ffbuf     = ws; ws += (size_t)BB * NN * FF;
  float* clsbuf    = ws; ws += (size_t)BB * DD;

  const int M = BB * NN;
  const float scale = 0.1020620726f;   // 1/sqrt(96)

  gauss_bias_kernel<<<dim3(NN * NN / 128, BB), 256, 0, stream>>>(
      dist, etype, gmeans, gstds, gmul, gbias, gp1w, gp1b, gp2w, gp2b, attn_bias);

  gather_embed_kernel<<<M, 256, 0, stream>>>(tok_emb, tokens, xbuf);
  ln_kernel<<<M, 256, 0, stream>>>(xbuf, emb_g, emb_b, xbuf);

  for (int l = 0; l < LL; ++l) {
    ln_kernel<<<M, 256, 0, stream>>>(xbuf, ln1g + l * DD, ln1b + l * DD, hbuf);
    gemm_kernel<<<dim3(DD / 128, M / 128), 256, 0, stream>>>(
        hbuf, DD, Wq + (size_t)l * DD * DD, DD, bq + l * DD, nullptr,
        pbuf, DD, M, DD, DD, 0);
    attn_score_kernel<<<dim3(2, 2, BB * HH), 256, 0, stream>>>(pbuf, attn_bias, sbuf, scale);
    softmax_kernel<<<BB * HH * NN, 256, 0, stream>>>(sbuf);
    attn_pv_kernel<<<dim3(2, BB * HH), 256, 0, stream>>>(sbuf, pbuf, obuf);
    gemm_kernel<<<dim3(DD / 128, M / 128), 256, 0, stream>>>(
        obuf, DD, Wo + (size_t)l * DD * DD, DD, bo + l * DD, xbuf,
        xbuf, DD, M, DD, DD, 0);
    ln_kernel<<<M, 256, 0, stream>>>(xbuf, ln2g + l * DD, ln2b + l * DD, hbuf);
    gemm_kernel<<<dim3(FF / 128, M / 128), 256, 0, stream>>>(
        hbuf, DD, W1 + (size_t)l * DD * FF, FF, b1 + l * FF, nullptr,
        ffbuf, FF, M, FF, DD, 1);
    gemm_kernel<<<dim3(DD / 128, M / 128), 256, 0, stream>>>(
        ffbuf, FF, W2 + (size_t)l * FF * DD, DD, b2 + l * DD, xbuf,
        xbuf, DD, M, DD, FF, 0);
  }

  ln_kernel<<<M, 256, 0, stream>>>(xbuf, fing, finb, hbuf);
  gemm_kernel<<<dim3(DD / 128, 1), 256, 0, stream>>>(
      hbuf, NN * DD, cw1, DD, cb1, nullptr, clsbuf, DD, BB, DD, DD, 1);
  gemm_kernel<<<dim3(1, 1), 256, 0, stream>>>(
      clsbuf, DD, cw2, 2, cb2, nullptr, out, 2, BB, 2, DD, 0);
}
